// MixtralMoE_37520834298349
// MI455X (gfx1250) — matrix operand order repaired and checked by reading, not compiled, not run on hardware
//
#include <hip/hip_runtime.h>
#include <hip/hip_bf16.h>

// ---------------------------------------------------------------------------
// MoE (Mixtral-style) top-2 routed SwiGLU FFN for gfx1250 (MI455X).
// Sparse routing + bf16 WMMA (v_wmma_f32_16x16x32_bf16) + fused FFN.
// Round 4: fix TDM builtin arity (6-arg toolchain form) and constant-imm
// s_wait_tensorcnt. TENSOR_LOAD_TO_LDS staging of weight tiles, double-
// buffered K-panels pipelined on TENSORcnt, async activation gather.
// ---------------------------------------------------------------------------

#define T_TOK 16384
#define H_DIM 1024
#define F_DIM 3584
#define N_EXP 8

typedef __attribute__((ext_vector_type(16))) __bf16         v16bf;
typedef __attribute__((ext_vector_type(8)))  float          v8f;
typedef __attribute__((ext_vector_type(16))) unsigned short u16x16;
typedef __attribute__((ext_vector_type(8)))  unsigned short u16x8;
typedef __attribute__((ext_vector_type(4)))  unsigned int   u32x4;
typedef __attribute__((ext_vector_type(8)))  int            i32x8;
typedef __attribute__((ext_vector_type(4)))  int            i32x4;

// fp32 -> bf16 round-to-nearest-even
__device__ __forceinline__ unsigned short f2bf(float f) {
  unsigned int u = __float_as_uint(f);
  u += 0x7FFFu + ((u >> 16) & 1u);
  return (unsigned short)(u >> 16);
}
__device__ __forceinline__ unsigned pack2(float a, float b) {
  return (unsigned)f2bf(a) | ((unsigned)f2bf(b) << 16);
}

// Async 16B copy global -> LDS (GLOBAL_LOAD_ASYNC_TO_LDS_B128, ASYNCcnt).
__device__ __forceinline__ void async_copy16(unsigned lds_off, const void* gptr) {
  asm volatile("global_load_async_to_lds_b128 %0, %1, off"
               :: "v"(lds_off), "v"(gptr) : "memory");
}
__device__ __forceinline__ void async_wait0() {
  asm volatile("s_wait_asynccnt 0" ::: "memory");
}

// ---------------------------------------------------------------------------
// TDM: one-instruction 2D tile DMA global -> LDS with LDS row padding.
// Descriptor per CDNA5 ISA (D# groups 0/1; groups 2/3
// zero => 2D). data_size=1 (2-byte elements). pad_interval/pad_amount codes:
// interval = 2^(code+1) dwords, amount = code+1 dwords.
// ---------------------------------------------------------------------------
__device__ __forceinline__ void tdm_load_2d(unsigned lds_off, const void* gaddr,
                                            unsigned tile_d0, unsigned tile_d1,
                                            unsigned tensor_d0, unsigned tensor_d1,
                                            unsigned d0_stride,
                                            unsigned pad_i_code, unsigned pad_a_code) {
  unsigned long long ga = (unsigned long long)gaddr;
  u32x4 g0;
  g0[0] = 1u;                                            // count=1 (valid user D#)
  g0[1] = lds_off;                                       // lds_addr (bytes)
  g0[2] = (unsigned)(ga & 0xFFFFFFFFu);                  // global_addr[31:0]
  g0[3] = (unsigned)((ga >> 32) & 0x1FFFFFFu) | (2u << 30);  // addr[56:32] | type=2
  i32x8 g1;
  g1[0] = (int)((1u << 16)                               // data_size = 1 (2B)
              | (1u << 20)                               // pad_enable
              | (pad_i_code << 22) | (pad_a_code << 25));
  g1[1] = (int)((tensor_d0 & 0xFFFFu) << 16);            // tensor_dim0[15:0] @ bits 63:48
  g1[2] = (int)(((tensor_d0 >> 16) & 0xFFFFu)            // tensor_dim0[31:16]
              | ((tensor_d1 & 0xFFFFu) << 16));          // tensor_dim1[15:0]
  g1[3] = (int)(((tensor_d1 >> 16) & 0xFFFFu)            // tensor_dim1[31:16]
              | ((tile_d0 & 0xFFFFu) << 16));            // tile_dim0
  g1[4] = (int)(tile_d1 & 0xFFFFu);                      // tile_dim1 (tile_dim2 = 0)
  g1[5] = (int)d0_stride;                                // tensor_dim0_stride[31:0]
  g1[6] = 0;                                             // stride hi / dim1_stride
  g1[7] = 0;
  i32x4 gz4  = {0, 0, 0, 0};
  i32x8 gz8  = {0, 0, 0, 0, 0, 0, 0, 0};
  // 6-arg toolchain form: (g0, g1, g2, g3, g4, cpol)
  __builtin_amdgcn_tensor_load_to_lds(g0, g1, gz4, gz4, gz8, 0);
}

// A-matrix operand (16x32 bf16): lanes 0-15 row M=L, K 0..7 & 16..23;
// lanes 16-31 row M=L-16, K 8..15 & 24..31. Two ds_load_b128 per lane.
__device__ __forceinline__ v16bf lds_load_A(const unsigned short* base, int stride,
                                            int row0, int k0, int lane) {
  const int r = lane & 15;
  const int h = (lane >> 4) & 1;
  const unsigned short* p = base + (row0 + r) * stride + k0 + h * 8;
  u16x8 lo = *(const u16x8*)p;
  u16x8 hi = *(const u16x8*)(p + 16);
  u16x16 v = __builtin_shufflevector(lo, hi, 0,1,2,3,4,5,6,7,8,9,10,11,12,13,14,15);
  return __builtin_bit_cast(v16bf, v);
}

// B-matrix operand (32x16 bf16), stored [n][k] in LDS: lanes 0-15 column N=L
// K 0..7 & 16..23; lanes 16-31 column N=L-16 K 8..15 & 24..31.
__device__ __forceinline__ v16bf lds_load_B(const unsigned short* base, int stride,
                                            int col0, int k0, int lane) {
  const int r = lane & 15;
  const int h = (lane >> 4) & 1;
  const unsigned short* p = base + (col0 + r) * stride + k0 + h * 8;
  u16x8 lo = *(const u16x8*)p;
  u16x8 hi = *(const u16x8*)(p + 16);
  u16x16 v = __builtin_shufflevector(lo, hi, 0,1,2,3,4,5,6,7,8,9,10,11,12,13,14,15);
  return __builtin_bit_cast(v16bf, v);
}

// ---------------------------------------------------------------------------
__global__ void zero_kernel(float4* __restrict__ out, int n4, int* __restrict__ counts) {
  int i = blockIdx.x * blockDim.x + threadIdx.x;
  if (i < n4) out[i] = make_float4(0.f, 0.f, 0.f, 0.f);
  if (i < N_EXP) counts[i] = 0;
}

__global__ __launch_bounds__(256)
void convert_kernel(const float4* __restrict__ src, uint2* __restrict__ dst, int n4) {
  int i = blockIdx.x * blockDim.x + threadIdx.x;
  if (i < n4) {
    float4 v = src[i];
    dst[i] = make_uint2(pack2(v.x, v.y), pack2(v.z, v.w));
  }
}

// ---------------------------------------------------------------------------
// Router: one wave32 per token, top-2 + renormalizing softmax, atomic scatter.
// ---------------------------------------------------------------------------
__global__ __launch_bounds__(256)
void router_kernel(const float* __restrict__ x, const float* __restrict__ gw,
                   int* __restrict__ counts, int* __restrict__ toks,
                   float* __restrict__ tokw) {
  const int tok  = (blockIdx.x * blockDim.x + threadIdx.x) >> 5;
  const int lane = threadIdx.x & 31;
  if (tok >= T_TOK) return;
  const float* xr = x + (size_t)tok * H_DIM;

  float logits[N_EXP];
  #pragma unroll
  for (int e = 0; e < N_EXP; ++e) {
    const float* g = gw + e * H_DIM;
    float s = 0.0f;
    for (int i = lane; i < H_DIM; i += 32) s += xr[i] * g[i];
    #pragma unroll
    for (int off = 16; off > 0; off >>= 1) s += __shfl_down(s, off, 32);
    logits[e] = s;
  }

  if (lane == 0) {
    int i1 = 0;
    #pragma unroll
    for (int e = 1; e < N_EXP; ++e) if (logits[e] > logits[i1]) i1 = e;
    int i2 = (i1 == 0) ? 1 : 0;
    #pragma unroll
    for (int e = 0; e < N_EXP; ++e) if (e != i1 && logits[e] > logits[i2]) i2 = e;
    float ez  = __expf(logits[i2] - logits[i1]);
    float inv = 1.0f / (1.0f + ez);
    int p1 = atomicAdd(&counts[i1], 1);
    toks[i1 * T_TOK + p1] = tok;  tokw[i1 * T_TOK + p1] = inv;
    int p2 = atomicAdd(&counts[i2], 1);
    toks[i2 * T_TOK + p2] = tok;  tokw[i2 * T_TOK + p2] = ez * inv;
  }
}

// ---------------------------------------------------------------------------
// Fused expert FFN. 512 threads (16 waves), one expert x 64 routed tokens.
// LDS (~297 KB of 320 KB):
//   xs   [64][1032]        bf16  gathered activations (resident)
//   wp   [2 buf][2 mat][32][264] bf16  w1/w3 K-panels, double buffered (TDM)
//   w2s  [1024][40]        bf16  w2 F-chunk, [n][f] rows (TDM)
//   gb/ub [64][33] f32, hb [64][40] bf16
// ---------------------------------------------------------------------------
#define XS_STR 1032
#define WP_STR 264
#define W2_STR 40
#define GU_STR 33
#define HB_STR 40
#define XS_OFF  0u
#define WP_BASE (64u * XS_STR * 2u)                          // 132096
#define WP_PANE (32u * WP_STR * 2u)                          // 16896
#define WP_OFF(buf, mat) (WP_BASE + (((buf) * 2u + (mat)) * WP_PANE))
#define W2S_OFF (WP_BASE + 4u * WP_PANE)                     // 199680
#define FFN_SMEM (64*XS_STR*2 + 4*32*WP_STR*2 + 1024*W2_STR*2 \
                  + 2*64*GU_STR*4 + 64*HB_STR*2 + 64*4 + 64*4)   // 304128

template <bool CACHED>
__global__ __launch_bounds__(512, 1)
void moe_ffn_kernel(const float* __restrict__ x,
                    const float* __restrict__ w1,
                    const float* __restrict__ w3,
                    const float* __restrict__ w2,
                    const unsigned short* __restrict__ xc,
                    const unsigned short* __restrict__ w1c,
                    const unsigned short* __restrict__ w3c,
                    const unsigned short* __restrict__ w2c,
                    const int* __restrict__ counts,
                    const int* __restrict__ toks,
                    const float* __restrict__ tokw,
                    float* __restrict__ out) {
  extern __shared__ unsigned char smem[];
  unsigned short* xs  = (unsigned short*)smem;              // [64][1032]
  unsigned short* wpB = xs  + 64 * XS_STR;                  // [4][32][264]
  unsigned short* w2s = wpB + 4 * 32 * WP_STR;              // [1024][40]
  float* gb  = (float*)(w2s + 1024 * W2_STR);               // [64][33]
  float* ub  = gb + 64 * GU_STR;                            // [64][33]
  unsigned short* hb = (unsigned short*)(ub + 64 * GU_STR); // [64][40]
  int*   tokL = (int*)(hb + 64 * HB_STR);                   // [64]
  float* twL  = (float*)(tokL + 64);                        // [64]

  const int e    = blockIdx.y;
  const int cnt  = counts[e];
  const int row0 = blockIdx.x * 64;
  if (row0 >= cnt) return;
  const int nrow = (cnt - row0 < 64) ? (cnt - row0) : 64;

  const int tid  = threadIdx.x;
  const int lane = tid & 31;
  const int wave = tid >> 5;

  if (tid < 64) {
    tokL[tid] = (tid < nrow) ? toks[e * T_TOK + row0 + tid] : 0;
    twL[tid]  = (tid < nrow) ? tokw[e * T_TOK + row0 + tid] : 0.0f;
  }
  __syncthreads();

  // Gather 64 token rows into LDS (resident across all F-chunks).
  if constexpr (CACHED) {
    for (int i = tid; i < 64 * 128; i += 512) {            // 16B bf16 chunks
      int r  = i >> 7;
      int c8 = (i & 127) << 3;
      async_copy16(XS_OFF + (unsigned)(r * XS_STR + c8) * 2u,
                   xc + (size_t)tokL[r] * H_DIM + c8);
    }
    async_wait0();
  } else {
    for (int i = tid; i < 64 * 256; i += 512) {
      int r = i >> 8;
      int c = (i & 255) << 2;
      float4 v = *(const float4*)(x + (size_t)tokL[r] * H_DIM + c);
      *(uint2*)(xs + r * XS_STR + c) = make_uint2(pack2(v.x, v.y), pack2(v.z, v.w));
    }
  }

  // GEMM1: waves 0-7 gate (w1), 8-15 up (w3); 16x16 tile each.
  const int tj = wave & 7;
  const int mt = tj >> 1;        // 0..3 row tile
  const int nt = tj & 1;         // 0..1 col tile
  const int wmat = (wave < 8) ? 0 : 1;
  // GEMM2: 4x4 wave grid, 16 rows x 256 cols each.
  const int wm = wave >> 2;
  const int wn = wave & 3;

  v8f acc2[16];
  const v8f vz = {0.f, 0.f, 0.f, 0.f, 0.f, 0.f, 0.f, 0.f};
  #pragma unroll
  for (int i = 0; i < 16; ++i) acc2[i] = vz;

  const size_t wbase = (size_t)e * F_DIM * H_DIM;

  #pragma unroll 1
  for (int f0 = 0; f0 < F_DIM; f0 += 32) {
    v8f accg = vz;
    if constexpr (CACHED) {
      if (wave == 0) {
        // w2 F-chunk [1024 n][32 f]: 64B rows padded to 80B (codes 3,3)
        tdm_load_2d(W2S_OFF, w2c + (size_t)e * H_DIM * F_DIM + f0,
                    32, 1024, F_DIM, H_DIM, F_DIM, 3, 3);
        // kp=0 panels [32 f][256 h]: 512B rows padded to 528B (codes 6,3)
        const unsigned short* p1 = w1c + wbase + (size_t)f0 * H_DIM;
        const unsigned short* p3 = w3c + wbase + (size_t)f0 * H_DIM;
        tdm_load_2d(WP_OFF(0, 0), p1, 256, 32, H_DIM, F_DIM, H_DIM, 6, 3);
        tdm_load_2d(WP_OFF(0, 1), p3, 256, 32, H_DIM, F_DIM, H_DIM, 6, 3);
      }
    }
    #pragma unroll 1
    for (int kp = 0; kp < 4; ++kp) {
      if (kp > 0) __syncthreads();       // buf[(kp+1)&1] consumers (kp-1) done
      if constexpr (CACHED) {
        if (wave == 0) {
          if (kp < 3) {
            int b = (kp + 1) & 1;
            const size_t po = wbase + (size_t)f0 * H_DIM + (size_t)(kp + 1) * 256;
            tdm_load_2d(WP_OFF(b, 0), w1c + po, 256, 32, H_DIM, F_DIM, H_DIM, 6, 3);
            tdm_load_2d(WP_OFF(b, 1), w3c + po, 256, 32, H_DIM, F_DIM, H_DIM, 6, 3);
            // in-order TDM completion: leave only the 2 newest outstanding
            __builtin_amdgcn_s_wait_tensorcnt(2);
          } else {
            __builtin_amdgcn_s_wait_tensorcnt(0);
          }
        }
      } else {
        for (int i = tid; i < 2 * 32 * 64; i += 512) {     // fp32 -> bf16, buf0
          int m  = i >> 11;
          int j  = i & 2047;
          int fr = j >> 6;
          int c  = (j & 63) << 2;
          const float* src = (m ? w3 : w1) + wbase + (size_t)(f0 + fr) * H_DIM + kp * 256 + c;
          float4 v = *(const float4*)src;
          *(uint2*)(wpB + m * 32 * WP_STR + fr * WP_STR + c) =
              make_uint2(pack2(v.x, v.y), pack2(v.z, v.w));
        }
      }
      __syncthreads();                   // publish panel buf[kp&1]
      const unsigned short* wk =
          wpB + ((CACHED ? ((kp & 1) * 2) : 0) + wmat) * (32 * WP_STR);
      #pragma unroll
      for (int ks = 0; ks < 8; ++ks) {
        v16bf a = lds_load_A(xs, XS_STR, mt * 16, kp * 256 + ks * 32, lane);
        v16bf b = lds_load_B(wk, WP_STR, nt * 16, ks * 32, lane);
        accg = __builtin_amdgcn_wmma_f32_16x16x32_bf16(
            false, a, false, b, (short)0, accg, false, false);
      }
    }
    // publish g/u tiles (C layout: vgpr r -> rows r / r+8, col = lane&15)
    {
      float* obuf = (wave < 8) ? gb : ub;
      #pragma unroll
      for (int r = 0; r < 8; ++r) {
        int row = mt * 16 + r + ((lane >> 4) << 3);
        obuf[row * GU_STR + nt * 16 + (lane & 15)] = accg[r];
      }
    }
    __syncthreads();

    // SwiGLU + routing weight -> hb (packed bf16 pairs)
    for (int i = tid; i < 64 * 16; i += 512) {
      int r = i >> 4, c = (i & 15) << 1;
      float g0 = gb[r * GU_STR + c], g1 = gb[r * GU_STR + c + 1];
      float u0 = ub[r * GU_STR + c], u1 = ub[r * GU_STR + c + 1];
      float s0 = g0 / (1.0f + __expf(-g0));
      float s1 = g1 / (1.0f + __expf(-g1));
      float w  = twL[r];
      *(unsigned*)(hb + r * HB_STR + c) = pack2(s0 * u0 * w, s1 * u1 * w);
    }
    if constexpr (!CACHED) {
      // stage w2 F-chunk [1024 n][32 f] (fp32 path)
      for (int i = tid; i < 1024 * 8; i += 512) {
        int n = i >> 3;
        int c = (i & 7) << 2;
        float4 v = *(const float4*)(w2 + (size_t)e * H_DIM * F_DIM + (size_t)n * F_DIM + f0 + c);
        *(uint2*)(w2s + n * W2_STR + c) = make_uint2(pack2(v.x, v.y), pack2(v.z, v.w));
      }
    }
    __syncthreads();    // hb (+w2s) visible; CACHED w2 TDM completed by kp waits

    // GEMM2: out[64,1024] += h[64,32] @ w2chunk^T
    v16bf a2 = lds_load_A(hb, HB_STR, wm * 16, 0, lane);
    #pragma unroll
    for (int ct = 0; ct < 16; ++ct) {
      v16bf b2 = lds_load_B(w2s, W2_STR, wn * 256 + ct * 16, 0, lane);
      acc2[ct] = __builtin_amdgcn_wmma_f32_16x16x32_bf16(
          false, a2, false, b2, (short)0, acc2[ct], false, false);
    }
    __syncthreads();    // hb/w2s consumed before next chunk's TDM reuses them
  }

  // epilogue: scatter-add output rows
  #pragma unroll
  for (int ct = 0; ct < 16; ++ct) {
    #pragma unroll
    for (int r = 0; r < 8; ++r) {
      int row = wm * 16 + r + ((lane >> 4) << 3);
      if (row < nrow) {
        atomicAdd(out + (size_t)tokL[row] * H_DIM + wn * 256 + ct * 16 + (lane & 15),
                  acc2[ct][r]);
      }
    }
  }
}

// ---------------------------------------------------------------------------
extern "C" void kernel_launch(void* const* d_in, const int* in_sizes, int n_in,
                              void* d_out, int out_size, void* d_ws, size_t ws_size,
                              hipStream_t stream) {
  (void)in_sizes; (void)n_in; (void)out_size;
  const float* x  = (const float*)d_in[0];
  const float* gw = (const float*)d_in[1];
  const float* w1 = (const float*)d_in[2];
  const float* w3 = (const float*)d_in[3];
  const float* w2 = (const float*)d_in[4];
  float* out = (float*)d_out;

  int*   counts = (int*)d_ws;
  int*   toks   = (int*)((char*)d_ws + 256);
  float* tokw   = (float*)((char*)d_ws + 256 + (size_t)N_EXP * T_TOK * sizeof(int));

  const size_t XC_OFF   = (size_t)2 << 20;
  const size_t XC_BYTES = (size_t)T_TOK * H_DIM * 2;
  const size_t W_ELEMS  = (size_t)N_EXP * F_DIM * H_DIM;
  const size_t W1C_OFF  = XC_OFF + XC_BYTES;
  const size_t W3C_OFF  = W1C_OFF + W_ELEMS * 2;
  const size_t W2C_OFF  = W3C_OFF + W_ELEMS * 2;
  const size_t WS_NEED  = W2C_OFF + W_ELEMS * 2;     // ~212 MB
  const bool cached = (ws_size >= WS_NEED);

  unsigned short* xc  = (unsigned short*)((char*)d_ws + XC_OFF);
  unsigned short* w1c = (unsigned short*)((char*)d_ws + W1C_OFF);
  unsigned short* w3c = (unsigned short*)((char*)d_ws + W3C_OFF);
  unsigned short* w2c = (unsigned short*)((char*)d_ws + W2C_OFF);

  {
    const int n4 = T_TOK * H_DIM / 4;
    zero_kernel<<<n4 / 256, 256, 0, stream>>>((float4*)out, n4, counts);
  }
  if (cached) {
    const int xn4 = T_TOK * H_DIM / 4;
    const int wn4 = (int)(W_ELEMS / 4);
    convert_kernel<<<(xn4 + 255) / 256, 256, 0, stream>>>((const float4*)x,  (uint2*)xc,  xn4);
    convert_kernel<<<(wn4 + 255) / 256, 256, 0, stream>>>((const float4*)w1, (uint2*)w1c, wn4);
    convert_kernel<<<(wn4 + 255) / 256, 256, 0, stream>>>((const float4*)w3, (uint2*)w3c, wn4);
    convert_kernel<<<(wn4 + 255) / 256, 256, 0, stream>>>((const float4*)w2, (uint2*)w2c, wn4);
  }
  router_kernel<<<(T_TOK * 32) / 256, 256, 0, stream>>>(x, gw, counts, toks, tokw);

  dim3 grid(T_TOK / 64, N_EXP);
  if (cached) {
    (void)hipFuncSetAttribute((const void*)moe_ffn_kernel<true>,
                              hipFuncAttributeMaxDynamicSharedMemorySize, FFN_SMEM);
    moe_ffn_kernel<true><<<grid, 512, FFN_SMEM, stream>>>(
        x, w1, w3, w2, xc, w1c, w3c, w2c, counts, toks, tokw, out);
  } else {
    (void)hipFuncSetAttribute((const void*)moe_ffn_kernel<false>,
                              hipFuncAttributeMaxDynamicSharedMemorySize, FFN_SMEM);
    moe_ffn_kernel<false><<<grid, 512, FFN_SMEM, stream>>>(
        x, w1, w3, w2, xc, w1c, w3c, w2c, counts, toks, tokw, out);
  }
}
